// AttnBlock_55070070669376
// MI455X (gfx1250) — matrix operand order repaired and checked by reading, not compiled, not run on hardware
//
#include <hip/hip_runtime.h>
#include <hip/hip_bf16.h>

// ---------------------------------------------------------------------------
// AttnBlock (SD-VAE style) for MI455X (gfx1250, wave32, WMMA).
// b=4, c=512, hw=4096. All GEMMs via v_wmma_f32_16x16x32_f16.
// Flash-attention: S never materialized in HBM; 32 queries per block;
// K/V tiles double-buffered in LDS via global_load_async_to_lds_b128
// (ASYNCcnt), overlapping the async copy of step i+1 with WMMAs of step i.
// ---------------------------------------------------------------------------

#define C_   512
#define HW_  4096
#define B_   4
#define NG_  32
#define CPG_ 16
#define SCALE_ 0.044194173824159216f   // 1/sqrt(512)

typedef __attribute__((ext_vector_type(16))) _Float16 v16h;
typedef __attribute__((ext_vector_type(8)))  _Float16 v8h;
typedef __attribute__((ext_vector_type(8)))  float    v8f;

// ---- WMMA helpers ----------------------------------------------------------

__device__ __forceinline__ v8f wmma_f16(v16h a, v16h b, v8f c) {
  // (neg_a, A, neg_b, B, c_mod, C, reuse_a, reuse_b)
  return __builtin_amdgcn_wmma_f32_16x16x32_f16(false, a, false, b, (short)0, c,
                                                false, false);
}

// A-matrix fragment (16x32 f16). Caller passes row base already offset by
// kkA = (lane<16 ? 0 : 8). Lane holds K {kkA..kkA+7} in v0-3 and
// K {16+kkA..23+kkA} in v4-7 as contiguous 16-bit pairs.
__device__ __forceinline__ v16h ldA(const _Float16* p) {
  v8h lo = *(const v8h*)(p);
  v8h hi = *(const v8h*)(p + 16);
  return __builtin_shufflevector(lo, hi, 0,1,2,3,4,5,6,7,8,9,10,11,12,13,14,15);
}

// B-matrix fragment (32x16 f16). Caller passes column base already offset by
// kkB = (lane<16 ? 0 : 8). Lane holds K {kkB..kkB+7} and K {16+kkB..23+kkB}.
__device__ __forceinline__ v16h ldB(const _Float16* p) {
  v8h lo = *(const v8h*)(p);
  v8h hi = *(const v8h*)(p + 16);
  return __builtin_shufflevector(lo, hi, 0,1,2,3,4,5,6,7,8,9,10,11,12,13,14,15);
}

// ---- Async copy to LDS (gfx1250, tracked by ASYNCcnt) ----------------------

__device__ __forceinline__ void async_ld_b128(unsigned lds_off, const void* gaddr) {
  asm volatile("global_load_async_to_lds_b128 %0, %1, off"
               :: "v"(lds_off), "v"((unsigned long long)(uintptr_t)gaddr)
               : "memory");
}
__device__ __forceinline__ void wait_async0() {
  asm volatile("s_wait_asynccnt 0x0" ::: "memory");
}

// ---- Kernel 1: f32 -> f16 weight convert -----------------------------------

__global__ void cvt_f16_k(const float* __restrict__ in, _Float16* __restrict__ out,
                          int n) {
  int i = blockIdx.x * blockDim.x + threadIdx.x;
  if (i < n) out[i] = (_Float16)in[i];
}

// ---- Kernel 2: GroupNorm, write transposed f16 [b][p][c] -------------------

__global__ __launch_bounds__(256)
void groupnorm_k(const float* __restrict__ x, const float* __restrict__ sc,
                 const float* __restrict__ bi, _Float16* __restrict__ hn_t) {
  const int b = blockIdx.y, g = blockIdx.x;
  const int tid = threadIdx.x;
  const float* base = x + (((size_t)b * C_) + (size_t)g * CPG_) * HW_; // 64K contiguous
  __shared__ float s_sum[256];
  __shared__ float s_sq[256];
  float s = 0.f, q = 0.f;
  for (int i = tid; i < CPG_ * HW_; i += 256) {
    float v = base[i];
    s += v; q += v * v;
  }
  s_sum[tid] = s; s_sq[tid] = q;
  __syncthreads();
  for (int o = 128; o > 0; o >>= 1) {
    if (tid < o) { s_sum[tid] += s_sum[tid + o]; s_sq[tid] += s_sq[tid + o]; }
    __syncthreads();
  }
  const float inv_n = 1.f / (float)(CPG_ * HW_);
  const float mean = s_sum[0] * inv_n;
  const float var  = s_sq[0] * inv_n - mean * mean;
  const float rinv = rsqrtf(var + 1e-6f);
  for (int i = tid; i < CPG_ * HW_; i += 256) {
    int cl = i >> 12;          // i / HW_
    int p  = i & (HW_ - 1);
    int c  = g * CPG_ + cl;
    float v = (base[i] - mean) * rinv * sc[c] + bi[c];
    hn_t[((size_t)b * HW_ + p) * C_ + c] = (_Float16)v;
  }
}

// ---- Kernel 3: channel GEMM  out[o,p] = sum_c W[o,c]*hn[c,p] + bias[o] -----

__global__ __launch_bounds__(256)
void gemm_qkv_k(const _Float16* __restrict__ hn_t, const _Float16* __restrict__ w16,
                const float* __restrict__ bias, _Float16* __restrict__ out,
                int transposed) {
  const int b    = blockIdx.z;
  const int p0   = blockIdx.x * 16;
  const int wave = threadIdx.x >> 5;
  const int lane = threadIdx.x & 31;
  const int ln   = lane & 15;
  const int kkA  = (lane < 16) ? 0 : 8;
  const int kkB  = (lane < 16) ? 0 : 8;
  const int mb   = (lane < 16) ? 0 : 8;
  const int o0   = blockIdx.y * 128 + wave * 16;

  const _Float16* arow = w16 + (size_t)(o0 + ln) * C_ + kkA;
  const _Float16* brow = hn_t + ((size_t)b * HW_ + p0 + ln) * C_ + kkB;

  v8f acc = {};
#pragma unroll
  for (int k0 = 0; k0 < C_; k0 += 32)
    acc = wmma_f16(ldA(arow + k0), ldB(brow + k0), acc);

  if (transposed) {
    v8h hv;
#pragma unroll
    for (int r = 0; r < 8; ++r) hv[r] = (_Float16)(acc[r] + bias[o0 + mb + r]);
    *(v8h*)(out + ((size_t)b * HW_ + p0 + ln) * C_ + o0 + mb) = hv;
  } else {
#pragma unroll
    for (int r = 0; r < 8; ++r)
      out[((size_t)b * C_ + o0 + mb + r) * HW_ + p0 + ln] =
          (_Float16)(acc[r] + bias[o0 + mb + r]);
  }
}

// ---- Kernel 4: flash attention with async LDS double-buffering -------------
// Block = 256 threads (8 waves), 32 queries (2 q-tiles) per block.
// Dynamic LDS: Kbuf[2][32][512] f16 (64KB) + Vbuf[2][512][32] f16 (64KB).
// Step i: issue async copies of step i+1's K/V tiles; compute S (WMMA from
// LDS), LDS-atomic reduce, online softmax, PV (WMMA from LDS);
// s_wait_asynccnt 0 + barrier.

__global__ __launch_bounds__(256)
void attn_k(const _Float16* __restrict__ q_t, const _Float16* __restrict__ k_t,
            const _Float16* __restrict__ v_, _Float16* __restrict__ ao_t) {
  const int b    = blockIdx.y;
  const int q0   = blockIdx.x * 32;
  const int tid  = threadIdx.x;
  const int wave = tid >> 5;
  const int lane = tid & 31;
  const int ln   = lane & 15;
  const int kkA  = (lane < 16) ? 0 : 8;
  const int kkB  = (lane < 16) ? 0 : 8;
  const int mb   = (lane < 16) ? 0 : 8;
  const int kg   = wave >> 2;         // S-phase key subtile (16 keys)
  const int ch0s = (wave & 3) * 128;  // S-phase channel group
  const int ch0p = wave * 64;         // PV-phase channel group

  extern __shared__ _Float16 dynbuf[];
  _Float16* Kb = dynbuf;                        // [2][32][512]
  _Float16* Vb = dynbuf + 2 * 32 * 512;         // [2][512][32]

  __shared__ float    Ssum[32][32];    // atomically reduced scores
  __shared__ _Float16 P[32][32];       // softmaxed probabilities (f16)
  __shared__ float    rowscale[32];
  __shared__ float    rowlinv[32];

  const size_t bhw = (size_t)b * HW_;
  const _Float16* kt_b = k_t + bhw * C_;               // [4096][512]
  const _Float16* v_b  = v_ + (size_t)b * C_ * HW_;    // [512][4096]

  // Zero Ssum once; softmax threads re-zero cells after consuming them.
  {
    float* sf = &Ssum[0][0];
#pragma unroll
    for (int i = 0; i < 4; ++i) sf[tid + 256 * i] = 0.f;
  }

  // Cooperative async staging of one K tile (32x512) and one V tile (512x32).
  auto stage_kv = [&](int kp, int buf) {
    const _Float16* kg0 = kt_b + (size_t)kp * C_;
    _Float16* kd = Kb + buf * (32 * 512);
#pragma unroll
    for (int j = 0; j < 8; ++j) {
      int i = tid + 256 * j;                 // 0..2047 chunks of 16B
      int row = i >> 6, c16 = i & 63;
      async_ld_b128((unsigned)(uintptr_t)(kd + row * 512 + c16 * 8),
                    kg0 + (size_t)row * C_ + c16 * 8);
    }
    const _Float16* vg0 = v_b + kp;
    _Float16* vd = Vb + buf * (512 * 32);
#pragma unroll
    for (int j = 0; j < 8; ++j) {
      int i = tid + 256 * j;                 // 0..2047 chunks of 16B
      int row = i >> 2, c4 = i & 3;
      async_ld_b128((unsigned)(uintptr_t)(vd + row * 32 + c4 * 8),
                    vg0 + (size_t)row * HW_ + c4 * 8);
    }
  };

  // Preload Q fragments (A-layout) for both q-tiles over S-phase channels.
  v16h Qa[2][4];
#pragma unroll
  for (int qt = 0; qt < 2; ++qt) {
    const _Float16* qrow =
        q_t + (bhw + q0 + qt * 16 + ln) * C_ + ch0s + kkA;
#pragma unroll
    for (int kc = 0; kc < 4; ++kc) Qa[qt][kc] = ldA(qrow + kc * 32);
  }

  v8f O[2][4] = {};
  float m_i = -3.0e38f, l_i = 0.f;

  // Prologue: stage step 0, wait, make Ssum-zero + tiles visible.
  stage_kv(0, 0);
  wait_async0();
  __syncthreads();

  for (int kp0 = 0; kp0 < HW_; kp0 += 32) {
    const int buf = (kp0 >> 5) & 1;
    if (kp0 + 32 < HW_) stage_kv(kp0 + 32, buf ^ 1);   // overlap with compute

    // --- partial S over this wave's (key subtile, channel group) ------------
    const _Float16* kb =
        Kb + buf * (32 * 512) + (kg * 16 + ln) * 512 + ch0s + kkB;
    v8f s[2] = {};
#pragma unroll
    for (int kc = 0; kc < 4; ++kc) {
      v16h Bk = ldB(kb + kc * 32);           // ds_load_b128 x2, reused by q-tiles
      s[0] = wmma_f16(Qa[0][kc], Bk, s[0]);
      s[1] = wmma_f16(Qa[1][kc], Bk, s[1]);
    }
#pragma unroll
    for (int qt = 0; qt < 2; ++qt)
#pragma unroll
      for (int r = 0; r < 8; ++r)
        atomicAdd(&Ssum[qt * 16 + mb + r][kg * 16 + ln], s[qt][r]);
    __syncthreads();

    // --- online softmax: 32 row threads, two-pass, re-zero ------------------
    if (tid < 32) {
      const int row = tid;
      float mx = m_i;
#pragma unroll
      for (int j = 0; j < 32; ++j)
        mx = fmaxf(mx, Ssum[row][j] * SCALE_);
      float rs = __expf(m_i - mx);
      float ss = 0.f;
#pragma unroll
      for (int j = 0; j < 32; ++j) {
        float p = __expf(Ssum[row][j] * SCALE_ - mx);
        Ssum[row][j] = 0.f;                  // ready for next step's atomics
        P[row][j] = (_Float16)p;
        ss += p;
      }
      l_i = l_i * rs + ss;
      m_i = mx;
      rowscale[row] = rs;
    }
    __syncthreads();

    // --- O = O*rescale + P · V over this wave's 64 channels -----------------
    v16h Pa[2];
    float rsc[2][8];
#pragma unroll
    for (int qt = 0; qt < 2; ++qt) {
      Pa[qt] = ldA(&P[qt * 16 + ln][0] + kkA);         // ds_load_b128 x2
#pragma unroll
      for (int r = 0; r < 8; ++r) rsc[qt][r] = rowscale[qt * 16 + mb + r];
    }
#pragma unroll
    for (int nt = 0; nt < 4; ++nt) {
      const _Float16* vb =
          Vb + buf * (512 * 32) + (ch0p + nt * 16 + ln) * 32 + kkB;
      v16h Bv = ldB(vb);                     // ds_load_b128 x2, reused by q-tiles
#pragma unroll
      for (int qt = 0; qt < 2; ++qt) {
#pragma unroll
        for (int r = 0; r < 8; ++r) O[qt][nt][r] *= rsc[qt][r];
        O[qt][nt] = wmma_f16(Pa[qt], Bv, O[qt][nt]);
      }
    }
    wait_async0();     // next tiles fully in LDS
    __syncthreads();   // ... and visible; P/rowscale/Ssum safe to reuse
  }

  if (tid < 32) rowlinv[tid] = (l_i > 0.f) ? (1.f / l_i) : 0.f;
  __syncthreads();

#pragma unroll
  for (int qt = 0; qt < 2; ++qt) {
    float li[8];
#pragma unroll
    for (int r = 0; r < 8; ++r) li[r] = rowlinv[qt * 16 + mb + r];
#pragma unroll
    for (int nt = 0; nt < 4; ++nt) {
      const int ch = ch0p + nt * 16 + ln;
#pragma unroll
      for (int r = 0; r < 8; ++r)
        ao_t[(bhw + q0 + qt * 16 + mb + r) * C_ + ch] =
            (_Float16)(O[qt][nt][r] * li[r]);
    }
  }
}

// ---- Kernel 5: proj GEMM + bias + residual, f32 out ------------------------

__global__ __launch_bounds__(256)
void gemm_proj_k(const _Float16* __restrict__ ao_t, const _Float16* __restrict__ wp16,
                 const float* __restrict__ bias, const float* __restrict__ x,
                 float* __restrict__ y) {
  const int b    = blockIdx.z;
  const int p0   = blockIdx.x * 16;
  const int wave = threadIdx.x >> 5;
  const int lane = threadIdx.x & 31;
  const int ln   = lane & 15;
  const int kkA  = (lane < 16) ? 0 : 8;
  const int kkB  = (lane < 16) ? 0 : 8;
  const int mb   = (lane < 16) ? 0 : 8;
  const int o0   = blockIdx.y * 128 + wave * 16;

  const _Float16* arow = wp16 + (size_t)(o0 + ln) * C_ + kkA;
  const _Float16* brow = ao_t + ((size_t)b * HW_ + p0 + ln) * C_ + kkB;

  v8f acc = {};
#pragma unroll
  for (int k0 = 0; k0 < C_; k0 += 32)
    acc = wmma_f16(ldA(arow + k0), ldB(brow + k0), acc);

#pragma unroll
  for (int r = 0; r < 8; ++r) {
    const size_t idx = ((size_t)b * C_ + o0 + mb + r) * HW_ + p0 + ln;
    y[idx] = x[idx] + bias[o0 + mb + r] + acc[r];
  }
}

// ---- Host launch -----------------------------------------------------------

extern "C" void kernel_launch(void* const* d_in, const int* in_sizes, int n_in,
                              void* d_out, int out_size, void* d_ws, size_t ws_size,
                              hipStream_t stream) {
  const float* x        = (const float*)d_in[0];
  const float* gn_scale = (const float*)d_in[1];
  const float* gn_bias  = (const float*)d_in[2];
  const float* wq       = (const float*)d_in[3];
  const float* bq       = (const float*)d_in[4];
  const float* wk       = (const float*)d_in[5];
  const float* bk       = (const float*)d_in[6];
  const float* wv       = (const float*)d_in[7];
  const float* bv       = (const float*)d_in[8];
  const float* wp       = (const float*)d_in[9];
  const float* bp       = (const float*)d_in[10];
  float* out = (float*)d_out;

  const size_t WEL = (size_t)C_ * C_;        // 262144 weight elements
  const size_t TEN = (size_t)B_ * HW_ * C_;  // 8388608 tensor elements

  _Float16* base = (_Float16*)d_ws;
  _Float16* wq16 = base;
  _Float16* wk16 = wq16 + WEL;
  _Float16* wv16 = wk16 + WEL;
  _Float16* wp16 = wv16 + WEL;
  _Float16* hn_t = wp16 + WEL;     // [b][p][c]
  _Float16* q_t  = hn_t + TEN;     // [b][p][c]
  _Float16* k_t  = q_t + TEN;      // [b][p][c]
  _Float16* v_   = k_t + TEN;      // [b][c][p]
  _Float16* ao_t = v_ + TEN;       // [b][p][c]

  // 1) weight converts
  {
    dim3 g((unsigned)((WEL + 255) / 256));
    cvt_f16_k<<<g, 256, 0, stream>>>(wq, wq16, (int)WEL);
    cvt_f16_k<<<g, 256, 0, stream>>>(wk, wk16, (int)WEL);
    cvt_f16_k<<<g, 256, 0, stream>>>(wv, wv16, (int)WEL);
    cvt_f16_k<<<g, 256, 0, stream>>>(wp, wp16, (int)WEL);
  }
  // 2) group norm -> hn_t
  groupnorm_k<<<dim3(NG_, B_), 256, 0, stream>>>(x, gn_scale, gn_bias, hn_t);

  // 3) q, k, v projections
  {
    dim3 g(HW_ / 16, C_ / 128, B_);
    gemm_qkv_k<<<g, 256, 0, stream>>>(hn_t, wq16, bq, q_t, 1);
    gemm_qkv_k<<<g, 256, 0, stream>>>(hn_t, wk16, bk, k_t, 1);
    gemm_qkv_k<<<g, 256, 0, stream>>>(hn_t, wv16, bv, v_, 0);
  }
  // 4) flash attention -> ao_t (32 queries/block; 128KB dynamic LDS for K/V)
  attn_k<<<dim3(HW_ / 32, B_), 256, 2 * (32 * 512 + 512 * 32) * sizeof(_Float16),
           stream>>>(q_t, k_t, v_, ao_t);

  // 5) proj + residual -> d_out (f32)
  gemm_proj_k<<<dim3(HW_ / 16, C_ / 128, B_), 256, 0, stream>>>(ao_t, wp16, bp, x, out);
}
